// CAAN_78323023610440
// MI455X (gfx1250) — matrix operand order repaired and checked by reading, not compiled, not run on hardware
//
#include <hip/hip_runtime.h>
#include <hip/hip_bf16.h>
#include <stdint.h>

typedef _Float16 h16;
typedef __attribute__((ext_vector_type(8)))  _Float16 v8h;
typedef __attribute__((ext_vector_type(16))) _Float16 v16h;
typedef __attribute__((ext_vector_type(8)))  float    v8f;
typedef __attribute__((ext_vector_type(4)))  int      v4i;

static constexpr int Bc = 4;
static constexpr int Nc = 2048;
static constexpr int Dc = 1024;
static constexpr int Mc = Bc * Nc; // 8192 flattened rows

#if __has_builtin(__builtin_amdgcn_global_load_async_to_lds_b128)
#define CAAN_HAVE_ASYNC 1
typedef __attribute__((address_space(1))) v4i gv4i;  // global int4
typedef __attribute__((address_space(3))) v4i lv4i;  // LDS int4
#endif

// 16-byte global -> LDS copy; async (ASYNCcnt) when the toolchain exposes it.
__device__ __forceinline__ void copy16(const h16* g, h16* l) {
#ifdef CAAN_HAVE_ASYNC
  __builtin_amdgcn_global_load_async_to_lds_b128((gv4i*)g, (lv4i*)l, 0, 0);
#else
  *(v8h*)l = *(const v8h*)g;
#endif
}

__device__ __forceinline__ void wait_copies() {
#ifdef CAAN_HAVE_ASYNC
#if __has_builtin(__builtin_amdgcn_s_wait_asynccnt)
  __builtin_amdgcn_s_wait_asynccnt(0);
#else
  asm volatile("s_wait_asynccnt 0" ::: "memory");
#endif
#endif
}

__device__ __forceinline__ v8h ld8(const h16* p) { return *(const v8h*)p; }

__device__ __forceinline__ v16h cat16(v8h lo, v8h hi) {
  return __builtin_shufflevector(lo, hi, 0,1,2,3,4,5,6,7,8,9,10,11,12,13,14,15);
}

// A-fragment 16x32 f16 (M x K). lane m = lane&15, half = lane>>4;
// element e in [0,8): K = 8*half + e ; e in [8,16): K = 16 + 8*half + (e-8).
__device__ __forceinline__ v16h loadA(const h16* base, int rowStride, int lane) {
  const int m = lane & 15, half = lane >> 4;
  const h16* p = base + (size_t)m * rowStride + half * 8;
  return cat16(ld8(p), ld8(p + 16));
}

// B-fragment 32x16 f16 (K x N) from src[n*rowStride + k].
// lane n = lane&15, half = lane>>4; element e in [0,8): K = 8*half + e ; e in [8,16): K = 16 + 8*half + (e-8).
__device__ __forceinline__ v16h loadB(const h16* base, int rowStride, int lane) {
  const int n = lane & 15, half = lane >> 4;
  const h16* p = base + (size_t)n * rowStride + half * 8;
  return cat16(ld8(p), ld8(p + 16));
}

__device__ __forceinline__ v8h cvt8(float4 a, float4 b) {
  v8h h;
  h[0] = (h16)a.x; h[1] = (h16)a.y; h[2] = (h16)a.z; h[3] = (h16)a.w;
  h[4] = (h16)b.x; h[5] = (h16)b.y; h[6] = (h16)b.z; h[7] = (h16)b.w;
  return h;
}

// ---------------------------------------------------------------------------
// Kernel 1: Y[m, dout] = (f16)( sum_k X[m,k]*W[dout,k] + bias[dout] )
// 64(M) x 128(Dout) tile per WG, 8 waves = 4(m) x 2(n), wave = 16x64 (4 acc).
// f32 -> f16 conversion staged through LDS with b128 loads/stores.
// ---------------------------------------------------------------------------
#define XS_STRIDE 40
__global__ void __launch_bounds__(256)
caan_proj_kernel(const float* __restrict__ X, const float* __restrict__ W,
                 const float* __restrict__ bias, h16* __restrict__ Y,
                 int M, int Dk, int Dout) {
  __shared__ alignas(16) h16 Xs[64 * XS_STRIDE];   // 64 rows x 32 (+8 pad)
  __shared__ alignas(16) h16 Ws[128 * XS_STRIDE];  // 128 rows x 32 (+8 pad)

  const int tid  = threadIdx.x;
  const int lane = tid & 31;
  const int wave = tid >> 5;
  const int mt = wave & 3;   // m sub-tile (16 rows)
  const int nh = wave >> 2;  // n half (64 cols)

  const int m0  = blockIdx.y * 64;
  const int n0g = blockIdx.x * 128;

  v8f acc[4] = {};

  for (int kk = 0; kk < Dk; kk += 32) {
    {  // stage X tile 64x32 (f32 -> f16), float4 loads, b128 LDS store
      const int row = tid >> 2;
      const int col = (tid & 3) * 8;
      const float4* xp = (const float4*)(X + (size_t)(m0 + row) * Dk + kk + col);
      *(v8h*)&Xs[row * XS_STRIDE + col] = cvt8(xp[0], xp[1]);
    }
    {  // stage W tile 128x32
      const int row = tid >> 1;
      const int col = (tid & 1) * 16;
      const float4* wp = (const float4*)(W + (size_t)(n0g + row) * Dk + kk + col);
      *(v8h*)&Ws[row * XS_STRIDE + col]     = cvt8(wp[0], wp[1]);
      *(v8h*)&Ws[row * XS_STRIDE + col + 8] = cvt8(wp[2], wp[3]);
    }
    __syncthreads();

    const v16h a = loadA(Xs + (mt * 16) * XS_STRIDE, XS_STRIDE, lane);
    #pragma unroll
    for (int j = 0; j < 4; ++j) {
      const v16h bf = loadB(Ws + (nh * 64 + j * 16) * XS_STRIDE, XS_STRIDE, lane);
      acc[j] = __builtin_amdgcn_wmma_f32_16x16x32_f16(
          false, a, false, bf, (short)0, acc[j], false, false);
    }
    __syncthreads();
  }

  const int lm = lane & 15, half = lane >> 4;
  #pragma unroll
  for (int j = 0; j < 4; ++j) {
    const int c = n0g + nh * 64 + j * 16 + lm;
    const float bj = bias[c];
    #pragma unroll
    for (int r = 0; r < 8; ++r) {
      const int row = m0 + mt * 16 + half * 8 + r;
      Y[(size_t)row * Dout + c] = (h16)(acc[j][r] + bj);
    }
  }
}

// ---------------------------------------------------------------------------
// Kernel 2: wv[k] = sum_d Ww[d]*Wv[d,k];  c0 = sum_d bv[d]*Ww[d]
// ---------------------------------------------------------------------------
__global__ void caan_wv_kernel(const float* __restrict__ Wv,
                               const float* __restrict__ Ww,
                               const float* __restrict__ bv,
                               float* __restrict__ wv, float* __restrict__ c0,
                               int D) {
  const int k = blockIdx.x * blockDim.x + threadIdx.x;
  if (k < D) {
    float acc = 0.f;
    for (int d = 0; d < D; ++d) acc += Ww[d] * Wv[(size_t)d * D + k];
    wv[k] = acc;
  }
  if (k == 0) {
    float c = 0.f;
    for (int d = 0; d < D; ++d) c += bv[d] * Ww[d];
    c0[0] = c;
  }
}

// ---------------------------------------------------------------------------
// Kernel 3: r[m] = x[m,:] . wv + c0     (r == V @ Ww^T, collapsed)
// ---------------------------------------------------------------------------
__global__ void __launch_bounds__(256)
caan_r_kernel(const float* __restrict__ X, const float* __restrict__ wv,
              const float* __restrict__ c0, float* __restrict__ r, int D) {
  __shared__ float red[256];
  const int row = blockIdx.x;
  const int tid = threadIdx.x;
  const float* xp = X + (size_t)row * D;
  float acc = 0.f;
  for (int k = tid; k < D; k += 256) acc += xp[k] * wv[k];
  red[tid] = acc;
  __syncthreads();
  for (int s = 128; s > 0; s >>= 1) {
    if (tid < s) red[tid] += red[tid + s];
    __syncthreads();
  }
  if (tid == 0) r[row] = red[0] + c0[0];
}

// ---------------------------------------------------------------------------
// Kernel 4: S[b,n,m] = scale * q[b,n,:] . k[b,m,:]
// 128x128 block tile, k-step 32, double-buffered async global->LDS staging.
// 8 waves; each wave computes 16 rows x 128 cols (8 accumulators, A reused).
// ---------------------------------------------------------------------------
__global__ void __launch_bounds__(256)
caan_scores_kernel(const h16* __restrict__ Q, const h16* __restrict__ K,
                   float* __restrict__ S, int N, int Dk) {
  // [buf][0]=Q tile, [buf][1]=K tile; 128 rows x 32 halfs (+8 pad) each
  __shared__ alignas(16) h16 Stage[2][2][128 * XS_STRIDE];  // 40 KB

  const int tid  = threadIdx.x;
  const int lane = tid & 31;
  const int wave = tid >> 5;
  const int b  = blockIdx.z;
  const int m0 = blockIdx.y * 128;
  const int n0 = blockIdx.x * 128;

  const h16* Qb = Q + (size_t)b * N * Dk;
  const h16* Kb = K + (size_t)b * N * Dk;
  float* Sb = S + (size_t)b * N * N;

  v8f acc[8] = {};

  // stage one 128x32 Q tile + 128x32 K tile: 512 16B-chunks each, 256 threads
  auto stage = [&](int buf, int kk) {
    #pragma unroll
    for (int c = tid; c < 512; c += 256) {
      const int row = c >> 2;
      const int sub = (c & 3) * 8;
      copy16(Qb + (size_t)(m0 + row) * Dk + kk + sub,
             &Stage[buf][0][row * XS_STRIDE + sub]);
      copy16(Kb + (size_t)(n0 + row) * Dk + kk + sub,
             &Stage[buf][1][row * XS_STRIDE + sub]);
    }
  };

  const int steps = Dk / 32;
  int buf = 0;
  stage(0, 0);

  for (int s = 0; s < steps; ++s) {
    wait_copies();
    __syncthreads();                      // tile(s) visible to all waves
    if (s + 1 < steps) stage(buf ^ 1, (s + 1) * 32);  // overlap next copy

    const v16h a = loadA(&Stage[buf][0][(wave * 16) * XS_STRIDE], XS_STRIDE, lane);
    #pragma unroll
    for (int j = 0; j < 8; ++j) {
      const v16h bf = loadB(&Stage[buf][1][(j * 16) * XS_STRIDE], XS_STRIDE, lane);
      acc[j] = __builtin_amdgcn_wmma_f32_16x16x32_f16(
          false, a, false, bf, (short)0, acc[j], false, false);
    }
    __syncthreads();                      // done reading buf before reuse
    buf ^= 1;
  }

  const float scale = 0.03125f; // 1/sqrt(1024)
  const int lm = lane & 15, half = lane >> 4;
  #pragma unroll
  for (int j = 0; j < 8; ++j) {
    const int col = n0 + j * 16 + lm;
    #pragma unroll
    for (int r = 0; r < 8; ++r) {
      const int row = m0 + wave * 16 + half * 8 + r;
      Sb[(size_t)row * N + col] = acc[j][r] * scale;
    }
  }
}

// ---------------------------------------------------------------------------
// Kernel 5: out[b,n] = (sum_m exp(S-mx)*r_b[m]) / (sum_m exp(S-mx)) + bw
// ---------------------------------------------------------------------------
__global__ void __launch_bounds__(256)
caan_final_kernel(const float* __restrict__ S, const float* __restrict__ r,
                  const float* __restrict__ bw, float* __restrict__ out, int N) {
  __shared__ float smax[256];
  __shared__ float ssum[256];
  __shared__ float swsm[256];

  const int n = blockIdx.x;
  const int b = blockIdx.y;
  const int tid = threadIdx.x;
  const float* sp = S + ((size_t)b * N + n) * N;
  const float* rb = r + (size_t)b * N;

  const int PER = N / 256; // 8
  float v[8];
  float lmax = -3.402823e38f;
  #pragma unroll
  for (int i = 0; i < PER; ++i) {
    v[i] = sp[i * 256 + tid];
    lmax = fmaxf(lmax, v[i]);
  }
  smax[tid] = lmax;
  __syncthreads();
  for (int s = 128; s > 0; s >>= 1) {
    if (tid < s) smax[tid] = fmaxf(smax[tid], smax[tid + s]);
    __syncthreads();
  }
  const float mx = smax[0];

  float psum = 0.f, pwsm = 0.f;
  #pragma unroll
  for (int i = 0; i < PER; ++i) {
    const float e = __expf(v[i] - mx);
    psum += e;
    pwsm += e * rb[i * 256 + tid];
  }
  ssum[tid] = psum;
  swsm[tid] = pwsm;
  __syncthreads();
  for (int s = 128; s > 0; s >>= 1) {
    if (tid < s) { ssum[tid] += ssum[tid + s]; swsm[tid] += swsm[tid + s]; }
    __syncthreads();
  }
  if (tid == 0) out[(size_t)b * N + n] = swsm[0] / ssum[0] + bw[0];
}

// ---------------------------------------------------------------------------
extern "C" void kernel_launch(void* const* d_in, const int* in_sizes, int n_in,
                              void* d_out, int out_size, void* d_ws, size_t ws_size,
                              hipStream_t stream) {
  const float* x  = (const float*)d_in[0];
  const float* Wq = (const float*)d_in[1];
  const float* bq = (const float*)d_in[2];
  const float* Wk = (const float*)d_in[3];
  const float* bk = (const float*)d_in[4];
  const float* Wv = (const float*)d_in[5];
  const float* bv = (const float*)d_in[6];
  const float* Ww = (const float*)d_in[7];
  const float* bw = (const float*)d_in[8];
  float* out = (float*)d_out;

  char* ws = (char*)d_ws;
  const size_t MB = (size_t)1 << 20;
  h16*   Qh  = (h16*)(ws + 0);                 // 16 MB
  h16*   Kh  = (h16*)(ws + 16 * MB);           // 16 MB
  float* wv  = (float*)(ws + 32 * MB);         // 4 KB
  float* c0  = wv + 1024;                      // 1 float
  float* rv  = wv + 2048;                      // 32 KB
  float* S   = (float*)(ws + 32 * MB + 65536); // 64 MB (lives in L2)

  // Q, K projections (WMMA GEMMs)
  caan_proj_kernel<<<dim3(Dc / 128, Mc / 64), dim3(256), 0, stream>>>(
      x, Wq, bq, Qh, Mc, Dc, Dc);
  caan_proj_kernel<<<dim3(Dc / 128, Mc / 64), dim3(256), 0, stream>>>(
      x, Wk, bk, Kh, Mc, Dc, Dc);

  // Collapsed V path: wv = Ww.Wv, c0 = bv.Ww, r = x.wv + c0
  caan_wv_kernel<<<dim3((Dc + 255) / 256), dim3(256), 0, stream>>>(
      Wv, Ww, bv, wv, c0, Dc);
  caan_r_kernel<<<dim3(Mc), dim3(256), 0, stream>>>(x, wv, c0, rv, Dc);

  // S = scale * Q K^T (WMMA, double-buffered async LDS staging)
  caan_scores_kernel<<<dim3(Nc / 128, Nc / 128, Bc), dim3(256), 0, stream>>>(
      Qh, Kh, S, Nc, Dc);

  // Fused softmax + weighted sum + bias
  caan_final_kernel<<<dim3(Nc, Bc), dim3(256), 0, stream>>>(S, rv, bw, out, Nc);
}
